// DifferentialBiomedCLIP_80573586473584
// MI455X (gfx1250) — matrix operand order repaired and checked by reading, not compiled, not run on hardware
//
#include <hip/hip_runtime.h>
#include <hip/hip_bf16.h>
#include <math.h>
#include <stdint.h>

// ---------------------------------------------------------------------------
// DifferentialBiomedCLIP block on MI455X (gfx1250, wave32, WMMA bf16 path)
// Async global->LDS staging for B tiles / Q tiles (ASYNCcnt), WMMA everywhere.
// ---------------------------------------------------------------------------

typedef __bf16 bf16;
typedef __attribute__((ext_vector_type(16))) __bf16 v16bf;
typedef __attribute__((ext_vector_type(8)))  __bf16 v8bf;
typedef __attribute__((ext_vector_type(8)))  float  v8f;

union AF { v16bf v; v8bf h[2]; };

__device__ __forceinline__ v8f wmma_bf16(v16bf a, v16bf b, v8f c) {
  // D = A(16x32) * B(32x16) + C(16x16 f32)
  return __builtin_amdgcn_wmma_f32_16x16x32_bf16(false, a, false, b, (short)0, c,
                                                 false, false);
}

// Async copy of 16B per lane from global to LDS (gfx1250, tracked by ASYNCcnt).
__device__ __forceinline__ void async_copy16(void* lds_dst, const void* gsrc) {
  unsigned ldsoff = (unsigned)(uintptr_t)lds_dst;            // low 32 bits = LDS offset
  unsigned long long ga = (unsigned long long)(uintptr_t)gsrc;
  asm volatile("global_load_async_to_lds_b128 %0, %1, off"
               :: "v"(ldsoff), "v"(ga) : "memory");
}
__device__ __forceinline__ void async_copy32(void* lds_dst, const void* gsrc) {
  unsigned ldsoff = (unsigned)(uintptr_t)lds_dst;
  unsigned long long ga = (unsigned long long)(uintptr_t)gsrc;
  asm volatile("global_load_async_to_lds_b128 %0, %1, off"
               :: "v"(ldsoff), "v"(ga) : "memory");
  asm volatile("global_load_async_to_lds_b128 %0, %1, off offset:16"
               :: "v"(ldsoff), "v"(ga) : "memory");
}
__device__ __forceinline__ void wait_async0() {
  asm volatile("s_wait_asynccnt 0x0" ::: "memory");
}

#define BATCH 128
#define SEQ   197
#define CDIM  768
#define HEADS 12
#define HD    64
#define MROWS (BATCH * SEQ)    // 25216 = 197 tiles of 128
#define NPAD  224              // key length padded to 14 tiles of 16
#define ATT_SCALE 0.125f       // 1/sqrt(64)

// ---------------------------------------------------------------------------
// Stage 0a: transpose-convert weight (C x C, row-major W[j][k]) -> bf16 WT[k][j]
// ---------------------------------------------------------------------------
__global__ __launch_bounds__(256) void k_transpose_w(const float* __restrict__ W,
                                                     bf16* __restrict__ WT) {
  int o = blockIdx.x * 256 + threadIdx.x;     // indexes WT linearly (589824)
  int k = o / CDIM;
  int j = o - k * CDIM;
  WT[o] = (bf16)W[j * CDIM + k];
}

// ---------------------------------------------------------------------------
// Stage 0b: lambda scalar
// ---------------------------------------------------------------------------
__global__ void k_lambda(const float* __restrict__ lq1, const float* __restrict__ lk1,
                         const float* __restrict__ lq2, const float* __restrict__ lk2,
                         const float* __restrict__ lam, float* __restrict__ lv) {
  if (threadIdx.x == 0) {
    float acc = 0.f;
    for (int e = 0; e < HEADS / 2; ++e) {
      float s1 = 0.f, s2 = 0.f;
      for (int d = 0; d < HD; ++d) {
        s1 += lq1[e * HD + d] * lk1[e * HD + d];
        s2 += lq2[e * HD + d] * lk2[e * HD + d];
      }
      acc += __expf(fminf(s1, 5.f)) - __expf(fminf(s2, 5.f));
    }
    *lv = acc / (float)(HEADS / 2) + lam[0];
  }
}

// ---------------------------------------------------------------------------
// Stage 1: QKV projection.  out[m][j] = sum_k X[m][k] * WT[k][j] + bias[j]
// Block tile 128x128, K-chunks of 32, 8 waves -> each wave 32x64 (2x4 WMMA tiles).
// B tile staged with async global->LDS; A tile converted f32->bf16 in VALU
// (overlaps the async latency). Epilogue scatters to [b][h][n][d] (bf16).
// ---------------------------------------------------------------------------
__global__ __launch_bounds__(256) void k_gemm_qkv(const float* __restrict__ X,
                                                  const bf16* __restrict__ WT,
                                                  const float* __restrict__ bias,
                                                  bf16* __restrict__ out) {
  __shared__ bf16 As[128][40];    // +8 pad: bank-spread for b128 frag loads
  __shared__ bf16 Bs[32][136];

  const int tid  = threadIdx.x;
  const int lane = tid & 31;
  const int w    = tid >> 5;
  const int wm   = w & 3;         // 4 waves along M
  const int wn   = w >> 2;        // 2 waves along N
  const int mt   = blockIdx.x;    // 197 tiles
  const int nt   = blockIdx.y;    // 6 tiles

  const v8f vz = {0.f, 0.f, 0.f, 0.f, 0.f, 0.f, 0.f, 0.f};
  v8f acc[2][4];
#pragma unroll
  for (int i = 0; i < 2; ++i)
#pragma unroll
    for (int j = 0; j < 4; ++j) acc[i][j] = vz;

  const int arow = tid >> 1;            // 0..127
  const int ac0  = (tid & 1) * 16;
  const int brow = tid >> 3;            // 0..31
  const int bc0  = (tid & 7) * 16;
  const float* xrow = X + (size_t)(mt * 128 + arow) * CDIM;

  for (int k0 = 0; k0 < CDIM; k0 += 32) {
    // ---- async-stage B tile (bf16, K-major): 32B per thread ----
    async_copy32(&Bs[brow][bc0],
                 WT + (size_t)(k0 + brow) * CDIM + nt * 128 + bc0);
    // ---- stage A tile (f32 -> bf16 on the fly), overlaps async B ----
    {
      const float4* src = (const float4*)(xrow + k0 + ac0);
      float tmp[16];
      *(float4*)(tmp + 0)  = src[0];
      *(float4*)(tmp + 4)  = src[1];
      *(float4*)(tmp + 8)  = src[2];
      *(float4*)(tmp + 12) = src[3];
      bf16* da = &As[arow][ac0];
#pragma unroll
      for (int i = 0; i < 16; ++i) da[i] = (bf16)tmp[i];
      if (k0 + 32 < CDIM) __builtin_prefetch(xrow + k0 + 32 + ac0, 0, 0);
    }
    wait_async0();
    __syncthreads();

    AF a[2];
#pragma unroll
    for (int i = 0; i < 2; ++i) {
      int row  = wm * 32 + i * 16 + (lane & 15);
      int koff = (lane >> 4) * 8;
      a[i].h[0] = *(const v8bf*)&As[row][koff];
      a[i].h[1] = *(const v8bf*)&As[row][16 + koff];
    }
#pragma unroll
    for (int j = 0; j < 4; ++j) {
      AF bfr;
      int n0 = wn * 64 + j * 16;
      for (int s = 0; s < 16; ++s)
        bfr.v[s] = Bs[(s & 7) + (lane >> 4) * 8 + (s >> 3) * 16][n0 + (lane & 15)];
#pragma unroll
      for (int i = 0; i < 2; ++i) acc[i][j] = wmma_bf16(a[i].v, bfr.v, acc[i][j]);
    }
    __syncthreads();
  }

  // ---- epilogue: bias + scatter to [b][h][n][d] ----
#pragma unroll
  for (int i = 0; i < 2; ++i)
#pragma unroll
    for (int j = 0; j < 4; ++j) {
      int n  = nt * 128 + wn * 64 + j * 16 + (lane & 15);
      int hh = n >> 6, d = n & 63;
      float bv = bias[n];
      int mbase = mt * 128 + wm * 32 + i * 16 + (lane >> 4) * 8;
#pragma unroll
      for (int r = 0; r < 8; ++r) {
        int m  = mbase + r;
        int bi = m / SEQ;
        int nn = m - bi * SEQ;
        out[(size_t)((bi * HEADS + hh) * SEQ + nn) * HD + d] = (bf16)(acc[i][j][r] + bv);
      }
    }
}

// ---------------------------------------------------------------------------
// Stage 2: attention per (b, h). Block: 256 threads; loops 7 row-tiles of 32.
// K^T staged in LDS; Q tile staged with async global->LDS; scores via WMMA;
// masked softmax (cols >= 197 zeroed); AV via WMMA with V B-frags streamed
// from global (lane-contiguous 32B per half-wave). Head h -> comb cols [h*64, h*64+64).
// ---------------------------------------------------------------------------
__global__ __launch_bounds__(256) void k_attn(const bf16* __restrict__ Q,
                                              const bf16* __restrict__ K,
                                              const bf16* __restrict__ V,
                                              bf16* __restrict__ comb) {
  __shared__ bf16  Kt[64][NPAD + 8];   // transposed keys: Kt[d][m]
  __shared__ bf16  Qs[32][HD + 8];
  __shared__ bf16  Sb[32][NPAD + 8];   // scores -> exp weights (unnormalized)
  __shared__ float red[32][8];
  __shared__ float rsum[32];

  const int tid  = threadIdx.x;
  const int lane = tid & 31;
  const int w    = tid >> 5;
  const int bh   = blockIdx.x;
  const int b    = bh / HEADS;
  const int h    = bh - b * HEADS;
  const size_t base = (size_t)bh * (SEQ * HD);
  const v8f vz = {0.f, 0.f, 0.f, 0.f, 0.f, 0.f, 0.f, 0.f};

  // ---- load K^T (zero-fill pad rows) ----
  if (tid < NPAD) {
    int m = tid;
    if (m < SEQ) {
      const bf16* kr = K + base + (size_t)m * HD;
#pragma unroll 8
      for (int d = 0; d < HD; ++d) Kt[d][m] = kr[d];
    } else {
#pragma unroll 8
      for (int d = 0; d < HD; ++d) Kt[d][m] = (bf16)0.f;
    }
  }
  __syncthreads();

  for (int t7 = 0; t7 < 7; ++t7) {
    const int row0 = t7 * 32;
    // ---- async-load Q tile (clamp pad rows; their output is discarded) ----
    {
      int r = tid >> 3, p = tid & 7;
      int gr = row0 + r;
      int sr = gr < SEQ ? gr : SEQ - 1;
      async_copy16(&Qs[r][p * 8], Q + base + (size_t)sr * HD + p * 8);
      wait_async0();
    }
    __syncthreads();

    // ---- scores: S = Q * K^T ----
    {
      const int wr = w & 1;       // 2 wave-rows of 16
      const int wc = w >> 1;      // 4 wave-cols, strided over 14 col tiles
      AF a[2];
#pragma unroll
      for (int kd = 0; kd < 2; ++kd) {
        int row  = wr * 16 + (lane & 15);
        int koff = kd * 32 + (lane >> 4) * 8;
        a[kd].h[0] = *(const v8bf*)&Qs[row][koff];
        a[kd].h[1] = *(const v8bf*)&Qs[row][koff + 16];
      }
      for (int jt = wc; jt < NPAD / 16; jt += 4) {
        v8f acc = vz;
#pragma unroll
        for (int kd = 0; kd < 2; ++kd) {
          AF bfr;
          for (int s = 0; s < 16; ++s)
            bfr.v[s] = Kt[kd * 32 + (s & 7) + (lane >> 4) * 8 + (s >> 3) * 16][jt * 16 + (lane & 15)];
          acc = wmma_bf16(a[kd].v, bfr.v, acc);
        }
        int mb  = wr * 16 + (lane >> 4) * 8;
        int col = jt * 16 + (lane & 15);
#pragma unroll
        for (int r = 0; r < 8; ++r) Sb[mb + r][col] = (bf16)acc[r];
      }
    }
    __syncthreads();

    // ---- masked softmax over 197 keys (row-wise, 8 threads/row) ----
    {
      int r = tid >> 3, p = tid & 7;
      float mx = -1e30f;
#pragma unroll 4
      for (int i = 0; i < 28; ++i) {
        int c = p * 28 + i;
        if (c < SEQ) mx = fmaxf(mx, (float)Sb[r][c]);
      }
      red[r][p] = mx;
      __syncthreads();
      float rmax = red[r][0];
#pragma unroll
      for (int i = 1; i < 8; ++i) rmax = fmaxf(rmax, red[r][i]);
      rmax *= ATT_SCALE;
      float s = 0.f;
#pragma unroll 4
      for (int i = 0; i < 28; ++i) {
        int c = p * 28 + i;
        if (c < SEQ) {
          float e = __expf(ATT_SCALE * (float)Sb[r][c] - rmax);
          Sb[r][c] = (bf16)e;
          s += e;
        } else {
          Sb[r][c] = (bf16)0.f;
        }
      }
      __syncthreads();               // everyone done reading red[][] as maxes
      red[r][p] = s;
      __syncthreads();
      if (p == 0) {
        float t = 0.f;
#pragma unroll
        for (int i = 0; i < 8; ++i) t += red[r][i];
        rsum[r] = t;
      }
    }
    __syncthreads();

    // ---- O = softmax(S) * V ; one 16x16 tile per wave (2x4) ----
    {
      const int mtile = w & 1;
      const int nt2   = w >> 1;
      v8f acc = vz;
      const int arow  = mtile * 16 + (lane & 15);
      const int koffb = (lane >> 4) * 8;
#pragma unroll
      for (int kc = 0; kc < NPAD / 32; ++kc) {
        AF a;
        a.h[0] = *(const v8bf*)&Sb[arow][kc * 32 + koffb];
        a.h[1] = *(const v8bf*)&Sb[arow][kc * 32 + 16 + koffb];
        AF bfr;
        for (int s = 0; s < 16; ++s) {
          int vrow = kc * 32 + (s & 7) + (lane >> 4) * 8 + (s >> 3) * 16;
          if (vrow > SEQ - 1) vrow = SEQ - 1;   // A is 0 there -> contributes 0
          bfr.v[s] = V[base + (size_t)vrow * HD + nt2 * 16 + (lane & 15)];
        }
        acc = wmma_bf16(a.v, bfr.v, acc);
      }
      int n   = nt2 * 16 + (lane & 15);
      int col = h * 64 + n;                   // concat layout == head order
      int mb  = mtile * 16 + (lane >> 4) * 8;
#pragma unroll
      for (int r = 0; r < 8; ++r) {
        int gr = row0 + mb + r;
        if (gr < SEQ)
          comb[(size_t)(b * SEQ + gr) * CDIM + col] = (bf16)(acc[r] / rsum[mb + r]);
      }
    }
    __syncthreads();
  }
}

// ---------------------------------------------------------------------------
// Stage 3: output projection with fused differential fixup.
// A-load: cols < 384 read comb - lv*comb[+384] (x1 - lv*x2); cols >= 384 pass x2.
// B tile staged async. Writes f32 result + bias to d_out.
// ---------------------------------------------------------------------------
__global__ __launch_bounds__(256) void k_gemm_proj(const bf16* __restrict__ Cmb,
                                                   const bf16* __restrict__ WT,
                                                   const float* __restrict__ bias,
                                                   const float* __restrict__ lvp,
                                                   float* __restrict__ out) {
  __shared__ bf16 As[128][40];
  __shared__ bf16 Bs[32][136];

  const int tid  = threadIdx.x;
  const int lane = tid & 31;
  const int w    = tid >> 5;
  const int wm   = w & 3;
  const int wn   = w >> 2;
  const int mt   = blockIdx.x;
  const int nt   = blockIdx.y;
  const float lv = *lvp;

  const v8f vz = {0.f, 0.f, 0.f, 0.f, 0.f, 0.f, 0.f, 0.f};
  v8f acc[2][4];
#pragma unroll
  for (int i = 0; i < 2; ++i)
#pragma unroll
    for (int j = 0; j < 4; ++j) acc[i][j] = vz;

  const int arow = tid >> 1;
  const int ac0  = (tid & 1) * 16;
  const int brow = tid >> 3;
  const int bc0  = (tid & 7) * 16;
  const bf16* crow = Cmb + (size_t)(mt * 128 + arow) * CDIM;

  for (int k0 = 0; k0 < CDIM; k0 += 32) {
    // ---- async-stage B tile ----
    async_copy32(&Bs[brow][bc0],
                 WT + (size_t)(k0 + brow) * CDIM + nt * 128 + bc0);
    // ---- stage A tile with differential fixup (overlaps async B) ----
    {
      bf16* da = &As[arow][ac0];
      const v8bf* s0 = (const v8bf*)(crow + k0 + ac0);
      if (k0 < 384) {                         // whole 32-chunk is in x1 region
        const v8bf* s1 = (const v8bf*)(crow + k0 + ac0 + 384);
        v8bf u0 = s0[0], u1 = s0[1], w0 = s1[0], w1 = s1[1];
#pragma unroll
        for (int i = 0; i < 8; ++i) da[i]     = (bf16)((float)u0[i] - lv * (float)w0[i]);
#pragma unroll
        for (int i = 0; i < 8; ++i) da[8 + i] = (bf16)((float)u1[i] - lv * (float)w1[i]);
      } else {
        *(v8bf*)&da[0] = s0[0];
        *(v8bf*)&da[8] = s0[1];
      }
    }
    wait_async0();
    __syncthreads();

    AF a[2];
#pragma unroll
    for (int i = 0; i < 2; ++i) {
      int row  = wm * 32 + i * 16 + (lane & 15);
      int koff = (lane >> 4) * 8;
      a[i].h[0] = *(const v8bf*)&As[row][koff];
      a[i].h[1] = *(const v8bf*)&As[row][16 + koff];
    }
#pragma unroll
    for (int j = 0; j < 4; ++j) {
      AF bfr;
      int n0 = wn * 64 + j * 16;
      for (int s = 0; s < 16; ++s)
        bfr.v[s] = Bs[(s & 7) + (lane >> 4) * 8 + (s >> 3) * 16][n0 + (lane & 15)];
#pragma unroll
      for (int i = 0; i < 2; ++i) acc[i][j] = wmma_bf16(a[i].v, bfr.v, acc[i][j]);
    }
    __syncthreads();
  }

#pragma unroll
  for (int i = 0; i < 2; ++i)
#pragma unroll
    for (int j = 0; j < 4; ++j) {
      int n = nt * 128 + wn * 64 + j * 16 + (lane & 15);
      float bv = bias[n];
      int mbase = mt * 128 + wm * 32 + i * 16 + (lane >> 4) * 8;
#pragma unroll
      for (int r = 0; r < 8; ++r) {
        int m = mbase + r;
        out[(size_t)m * CDIM + n] = acc[i][j][r] + bv;
      }
    }
}

// ---------------------------------------------------------------------------
// Host launcher
// ---------------------------------------------------------------------------
extern "C" void kernel_launch(void* const* d_in, const int* in_sizes, int n_in,
                              void* d_out, int out_size, void* d_ws, size_t ws_size,
                              hipStream_t stream) {
  (void)in_sizes; (void)n_in; (void)out_size; (void)ws_size;
  const float* x   = (const float*)d_in[0];
  const float* Wq  = (const float*)d_in[1];
  const float* bq  = (const float*)d_in[2];
  const float* Wk  = (const float*)d_in[3];
  const float* bk  = (const float*)d_in[4];
  const float* Wv  = (const float*)d_in[5];
  const float* bv  = (const float*)d_in[6];
  const float* Wp  = (const float*)d_in[7];
  const float* bp  = (const float*)d_in[8];
  const float* lq1 = (const float*)d_in[9];
  const float* lk1 = (const float*)d_in[10];
  const float* lq2 = (const float*)d_in[11];
  const float* lk2 = (const float*)d_in[12];
  const float* lam = (const float*)d_in[14];   // d_in[13] = num_heads (unused)

  char* ws = (char*)d_ws;
  size_t off = 0;
  auto carve = [&](size_t bytes) -> char* {
    char* p = ws + off;
    off += (bytes + 255) & ~(size_t)255;
    return p;
  };
  const size_t CC  = (size_t)CDIM * CDIM;
  const size_t QKB = (size_t)BATCH * HEADS * SEQ * HD * sizeof(bf16);

  float* lv  = (float*)carve(sizeof(float));
  bf16* wqT  = (bf16*)carve(CC * sizeof(bf16));
  bf16* wkT  = (bf16*)carve(CC * sizeof(bf16));
  bf16* wvT  = (bf16*)carve(CC * sizeof(bf16));
  bf16* wpT  = (bf16*)carve(CC * sizeof(bf16));
  bf16* qb   = (bf16*)carve(QKB);
  bf16* kb   = (bf16*)carve(QKB);
  bf16* vb   = (bf16*)carve(QKB);
  bf16* comb = (bf16*)carve((size_t)MROWS * CDIM * sizeof(bf16));

  const int tgrid = (int)(CC / 256);            // 2304
  k_transpose_w<<<tgrid, 256, 0, stream>>>(Wq, wqT);
  k_transpose_w<<<tgrid, 256, 0, stream>>>(Wk, wkT);
  k_transpose_w<<<tgrid, 256, 0, stream>>>(Wv, wvT);
  k_transpose_w<<<tgrid, 256, 0, stream>>>(Wp, wpT);
  k_lambda<<<1, 64, 0, stream>>>(lq1, lk1, lq2, lk2, lam, lv);

  dim3 ggrid(MROWS / 128, CDIM / 128);          // (197, 6)
  k_gemm_qkv<<<ggrid, 256, 0, stream>>>(x, wqT, bq, qb);
  k_gemm_qkv<<<ggrid, 256, 0, stream>>>(x, wkT, bk, kb);
  k_gemm_qkv<<<ggrid, 256, 0, stream>>>(x, wvT, bv, vb);

  k_attn<<<BATCH * HEADS, 256, 0, stream>>>(qb, kb, vb, comb);

  k_gemm_proj<<<ggrid, 256, 0, stream>>>(comb, wpT, bp, lv, (float*)d_out);
}
